// BiasedAxialAttention_29583734735261
// MI455X (gfx1250) — matrix operand order repaired and checked by reading, not compiled, not run on hardware
//
#include <hip/hip_runtime.h>
#include <math.h>

#define DEVINL __device__ __forceinline__
#define AS1 __attribute__((address_space(1)))

typedef __attribute__((ext_vector_type(16))) _Float16 v16h;
typedef __attribute__((ext_vector_type(8)))  float    v8f;
typedef __attribute__((ext_vector_type(4)))  unsigned int u32x4;
typedef __attribute__((ext_vector_type(4)))  float    f32x4;

constexpr int L  = 384;
constexpr int DP = 128;
constexpr int H  = 4;
constexpr int DH = 32;
constexpr int KC = L * DH;                       // 12288, contraction length for scores
constexpr float SCALING = 0.17677669529663687f;  // 1/sqrt(DH)

// ---- explicit global (address_space(1)) accessors: force global_load/store ----
DEVINL u32x4  ldg_u4(const void* p)            { return *(const AS1 u32x4*)(unsigned long long)p; }
DEVINL void   stg_u4(void* p, u32x4 v)         { *(AS1 u32x4*)(unsigned long long)p = v; }
DEVINL f32x4  ldg_f4(const float* p)           { return *(const AS1 f32x4*)(unsigned long long)p; }
DEVINL float  ldg_f(const float* p)            { return *(const AS1 float*)(unsigned long long)p; }
DEVINL void   stg_f(float* p, float v)         { *(AS1 float*)(unsigned long long)p = v; }
DEVINL _Float16 ldg_h(const _Float16* p)       { return *(const AS1 _Float16*)(unsigned long long)p; }
DEVINL void   stg_h(_Float16* p, _Float16 v)   { *(AS1 _Float16*)(unsigned long long)p = v; }

union FragU { v16h v; u32x4 q[2]; };

// A fragment (16x32 f16, row-major, ld in f16 units) from GLOBAL.
// Lanes 0-15: row m=lane, K=0..7 / 16..23; lanes 16-31: K=8..15 / 24..31.
DEVINL v16h load_frag_a_g(const _Float16* base, int ld, int lane) {
  const int m = lane & 15, sel = lane >> 4;
  const _Float16* r = base + (size_t)m * ld + sel * 8;
  FragU f;
  f.q[0] = ldg_u4(r);
  f.q[1] = ldg_u4(r + 16);
  return f.v;
}

// B fragment (32x16 f16 given as B^T rows, ld in f16 units) from GLOBAL.
DEVINL v16h load_frag_b_g(const _Float16* base, int ld, int lane) {
  const int n = lane & 15, sel = lane >> 4;
  const _Float16* r = base + (size_t)n * ld + sel * 8;
  FragU f;
  f.q[0] = ldg_u4(r);
  f.q[1] = ldg_u4(r + 16);
  return f.v;
}

// A fragment from LDS (direct shared-array access -> ds_load_b128)
DEVINL v16h load_frag_a_lds(const _Float16* base, int ld, int lane) {
  const int m = lane & 15, sel = lane >> 4;
  const _Float16* r = base + m * ld + sel * 8;
  FragU f;
  f.q[0] = *(const u32x4*)(r);
  f.q[1] = *(const u32x4*)(r + 16);
  return f.v;
}

DEVINL v8f wmma16(v16h a, v16h b, v8f c) {
  return __builtin_amdgcn_wmma_f32_16x16x32_f16(false, a, false, b, (short)0, c, false, false);
}

// ---------------- weight conversion ----------------
__global__ void k_f2h(const float* __restrict__ s, _Float16* __restrict__ d, int nelem) {
  int i = blockIdx.x * blockDim.x + threadIdx.x;
  if (i < nelem) stg_h(d + i, (_Float16)ldg_f(s + i));
}

__global__ void k_wbpad(const float* __restrict__ s, _Float16* __restrict__ d) {
  int i = blockIdx.x * blockDim.x + threadIdx.x;   // 16x128 padded Wb
  if (i < 16 * DP) {
    int r = i >> 7, c = i & 127;
    stg_h(d + i, (r < H) ? (_Float16)ldg_f(s + r * DP + c) : (_Float16)0.0f);
  }
}

// shared helper: compute one 16x16 projection tile (A from LDS, B^T = weight rows)
DEVINL v8f proj_tile(const _Float16* Atile0, const _Float16* W, int mi, int nt, int lane) {
  v8f acc = {};
#pragma unroll
  for (int ks = 0; ks < 4; ++ks) {
    v16h a  = load_frag_a_lds(Atile0 + (mi * 16) * 136 + ks * 32, 136, lane);
    v16h bf = load_frag_b_g(W + (size_t)nt * 16 * DP + ks * 32, DP, lane);
    acc = wmma16(a, bf, acc);
  }
  return acc;
}

// ---------------- kernel 1: LN(pair^T) + Q/K/V/Gate projections ----------------
// Block: 256 threads, 64 rows (n fixed, i = i0..i0+63).  p[n][i][c] = pair[i][n][c].
__global__ void __launch_bounds__(256)
k_ln_proj_pair(const float* __restrict__ pair,
               const float* __restrict__ lng, const float* __restrict__ lnb,
               const _Float16* __restrict__ wq, const _Float16* __restrict__ wk,
               const _Float16* __restrict__ wv, const _Float16* __restrict__ wg,
               const float* __restrict__ bg,
               _Float16* __restrict__ Qt, _Float16* __restrict__ Kt,
               _Float16* __restrict__ Vt, _Float16* __restrict__ Gate)
{
  __shared__ _Float16 Atile[64][136];
  __shared__ _Float16 Vtile[64][136];
  __shared__ float redS[64][4];
  __shared__ float redQ[64][4];

  const int t   = threadIdx.x;
  const int blk = blockIdx.x;          // 2304 = 384 n * 6 i-tiles
  const int n   = blk / 6;
  const int i0  = (blk % 6) * 64;

  // Phase 1: load + LayerNorm (4 threads per row, 32 channels each)
  {
    const int row = t >> 2, part = t & 3, c0 = part * 32;
    const float* src = pair + ((size_t)(i0 + row) * L + n) * DP + c0;
    float x[32];
#pragma unroll
    for (int q = 0; q < 8; ++q) {
      f32x4 f = ldg_f4(src + 4 * q);
      x[4*q] = f.x; x[4*q+1] = f.y; x[4*q+2] = f.z; x[4*q+3] = f.w;
    }
    float s = 0.f, s2 = 0.f;
#pragma unroll
    for (int c = 0; c < 32; ++c) { s += x[c]; s2 += x[c] * x[c]; }
    redS[row][part] = s; redQ[row][part] = s2;
    __syncthreads();
    float fs = redS[row][0] + redS[row][1] + redS[row][2] + redS[row][3];
    float fq = redQ[row][0] + redQ[row][1] + redQ[row][2] + redQ[row][3];
    float mean = fs * (1.0f / DP);
    float var  = fq * (1.0f / DP) - mean * mean;
    float rstd = rsqrtf(var + 1e-5f);
#pragma unroll
    for (int c = 0; c < 32; ++c) {
      float v = (x[c] - mean) * rstd * ldg_f(lng + c0 + c) + ldg_f(lnb + c0 + c);
      Atile[row][c0 + c] = (_Float16)v;
    }
  }
  __syncthreads();

  // Phase 2: four 64x128x128 projections via WMMA (straight-line, no runtime selects)
  const int w = t >> 5, lane = t & 31;
  const int mi = w & 3;
  const int ntg = (w >> 2) * 4;
  const int nn = lane & 15, sel = lane >> 4;
  const _Float16* A0 = &Atile[0][0];

  // ---- Q: scaled, layout Qt[h][i][n][k] ----
#pragma unroll
  for (int tt = 0; tt < 4; ++tt) {
    const int nt = ntg + tt;
    v8f acc = proj_tile(A0, wq, mi, nt, lane);
    const int o = nt * 16 + nn;
    const int h = o >> 5, k = o & 31;
#pragma unroll
    for (int v = 0; v < 8; ++v) {
      const int irow = i0 + mi * 16 + v + 8 * sel;
      stg_h(Qt + (((size_t)h * L + irow) * L + n) * DH + k, (_Float16)(acc[v] * SCALING));
    }
  }
  // ---- K: /L, layout Kt[h][j][n][k] ----
#pragma unroll
  for (int tt = 0; tt < 4; ++tt) {
    const int nt = ntg + tt;
    v8f acc = proj_tile(A0, wk, mi, nt, lane);
    const int o = nt * 16 + nn;
    const int h = o >> 5, k = o & 31;
#pragma unroll
    for (int v = 0; v < 8; ++v) {
      const int irow = i0 + mi * 16 + v + 8 * sel;
      stg_h(Kt + (((size_t)h * L + irow) * L + n) * DH + k, (_Float16)(acc[v] * (1.0f / L)));
    }
  }
  // ---- V: stage to LDS for transposed store ----
#pragma unroll
  for (int tt = 0; tt < 4; ++tt) {
    const int nt = ntg + tt;
    v8f acc = proj_tile(A0, wv, mi, nt, lane);
    const int o = nt * 16 + nn;
#pragma unroll
    for (int v = 0; v < 8; ++v) {
      Vtile[mi * 16 + v + 8 * sel][o] = (_Float16)acc[v];
    }
  }
  // ---- Gate: sigmoid(. + bg), layout Gate[n*L+i][hd] ----
#pragma unroll
  for (int tt = 0; tt < 4; ++tt) {
    const int nt = ntg + tt;
    v8f acc = proj_tile(A0, wg, mi, nt, lane);
    const int o = nt * 16 + nn;
    const float bgo = ldg_f(bg + o);
#pragma unroll
    for (int v = 0; v < 8; ++v) {
      const int irow = i0 + mi * 16 + v + 8 * sel;
      float g = 1.0f / (1.0f + expf(-(acc[v] + bgo)));
      stg_h(Gate + ((size_t)n * L + irow) * DP + o, (_Float16)g);
    }
  }
  __syncthreads();

  // Phase 3: transposed V store: Vt[((h*L + n)*DH + d)*L + j], j contiguous
  {
    const int hd = t >> 1, half = t & 1;
    const int h = hd >> 5, d = hd & 31;
    union { _Float16 f[32]; u32x4 q[4]; } buf;
#pragma unroll
    for (int jj = 0; jj < 32; ++jj) buf.f[jj] = Vtile[half * 32 + jj][hd];
    _Float16* dst = Vt + (((size_t)h * L + n) * DH + d) * L + i0 + half * 32;
#pragma unroll
    for (int q = 0; q < 4; ++q) stg_u4(dst + 8 * q, buf.q[q]);
  }
}

// ---------------- kernel 1b: LN(bias^T) + Wb projection ----------------
// Rows (i fixed, j = j0..j0+63).  bsrc[i][j][c] = bias[j][i][c].
__global__ void __launch_bounds__(256)
k_ln_proj_bias(const float* __restrict__ bias,
               const float* __restrict__ lng, const float* __restrict__ lnb,
               const _Float16* __restrict__ wb,   // padded [16][128]
               float* __restrict__ biasP)         // [h][i][j]
{
  __shared__ _Float16 Atile[64][136];
  __shared__ float redS[64][4];
  __shared__ float redQ[64][4];

  const int t   = threadIdx.x;
  const int blk = blockIdx.x;
  const int i   = blk / 6;
  const int j0  = (blk % 6) * 64;

  {
    const int row = t >> 2, part = t & 3, c0 = part * 32;
    const float* src = bias + ((size_t)(j0 + row) * L + i) * DP + c0;
    float x[32];
#pragma unroll
    for (int q = 0; q < 8; ++q) {
      f32x4 f = ldg_f4(src + 4 * q);
      x[4*q] = f.x; x[4*q+1] = f.y; x[4*q+2] = f.z; x[4*q+3] = f.w;
    }
    float s = 0.f, s2 = 0.f;
#pragma unroll
    for (int c = 0; c < 32; ++c) { s += x[c]; s2 += x[c] * x[c]; }
    redS[row][part] = s; redQ[row][part] = s2;
    __syncthreads();
    float fs = redS[row][0] + redS[row][1] + redS[row][2] + redS[row][3];
    float fq = redQ[row][0] + redQ[row][1] + redQ[row][2] + redQ[row][3];
    float mean = fs * (1.0f / DP);
    float var  = fq * (1.0f / DP) - mean * mean;
    float rstd = rsqrtf(var + 1e-5f);
#pragma unroll
    for (int c = 0; c < 32; ++c) {
      float v = (x[c] - mean) * rstd * ldg_f(lng + c0 + c) + ldg_f(lnb + c0 + c);
      Atile[row][c0 + c] = (_Float16)v;
    }
  }
  __syncthreads();

  const int w = t >> 5, lane = t & 31;
  if (w < 4) {                       // wave-uniform: EXEC all-1 inside
    const int mi = w;
    const int nn = lane & 15, sel = lane >> 4;
    v8f acc = proj_tile(&Atile[0][0], wb, mi, 0, lane);
    if (nn < H) {
      const int h = nn;
#pragma unroll
      for (int v = 0; v < 8; ++v) {
        const int jrow = j0 + mi * 16 + v + 8 * sel;
        stg_f(biasP + ((size_t)h * L + i) * L + jrow, acc[v]);
      }
    }
  }
}

// ---------------- kernel 2: attention scores  attn[h,i,j] = Q·K + bias ----------------
__global__ void __launch_bounds__(256)
k_scores(const _Float16* __restrict__ Qt, const _Float16* __restrict__ Kt,
         const float* __restrict__ biasP, float* __restrict__ attnS)
{
  const int h  = blockIdx.y;
  const int i0 = (blockIdx.x / 6) * 64, j0 = (blockIdx.x % 6) * 64;
  const int t = threadIdx.x, w = t >> 5, lane = t & 31;
  const int mi = w & 3, njb = (w >> 2) * 2;
  const int nn = lane & 15, sel = lane >> 4;

  const _Float16* Abase = Qt + (size_t)h * L * KC + (size_t)(i0 + mi * 16) * KC;
  const _Float16* B0 = Kt + (size_t)h * L * KC + (size_t)(j0 + njb * 16) * KC;
  const _Float16* B1 = B0 + (size_t)16 * KC;

  v8f acc0 = {}, acc1 = {};
#pragma unroll 4
  for (int kk = 0; kk < KC / 32; ++kk) {
    const int c0 = kk * 32;
    v16h a  = load_frag_a_g(Abase + c0, KC, lane);
    v16h b0 = load_frag_b_g(B0 + c0, KC, lane);
    acc0 = wmma16(a, b0, acc0);
    v16h b1 = load_frag_b_g(B1 + c0, KC, lane);
    acc1 = wmma16(a, b1, acc1);
  }
#pragma unroll
  for (int v = 0; v < 8; ++v) {
    const int ii = i0 + mi * 16 + v + 8 * sel;
    const size_t rb = ((size_t)h * L + ii) * L;
    const int jj0 = j0 + njb * 16 + nn;
    stg_f(attnS + rb + jj0,      acc0[v] + ldg_f(biasP + rb + jj0));
    stg_f(attnS + rb + jj0 + 16, acc1[v] + ldg_f(biasP + rb + jj0 + 16));
  }
}

// ---------------- kernel 3: row softmax over j, emit f16 weights ----------------
__global__ void __launch_bounds__(128)
k_softmax(const float* __restrict__ attnS, _Float16* __restrict__ attnW)
{
  __shared__ float red[128];
  const int row = blockIdx.x;          // h*L + i
  const int t = threadIdx.x;
  const size_t base = (size_t)row * L;
  float x0 = ldg_f(attnS + base + t);
  float x1 = ldg_f(attnS + base + t + 128);
  float x2 = ldg_f(attnS + base + t + 256);
  float m = fmaxf(x0, fmaxf(x1, x2));
  red[t] = m; __syncthreads();
  for (int s = 64; s > 0; s >>= 1) { if (t < s) red[t] = fmaxf(red[t], red[t + s]); __syncthreads(); }
  m = red[0]; __syncthreads();
  float e0 = expf(x0 - m), e1 = expf(x1 - m), e2 = expf(x2 - m);
  red[t] = e0 + e1 + e2; __syncthreads();
  for (int s = 64; s > 0; s >>= 1) { if (t < s) red[t] += red[t + s]; __syncthreads(); }
  float inv = 1.0f / red[0];
  stg_h(attnW + base + t,       (_Float16)(e0 * inv));
  stg_h(attnW + base + t + 128, (_Float16)(e1 * inv));
  stg_h(attnW + base + t + 256, (_Float16)(e2 * inv));
}

// ---------------- kernel 4: out[n,i,h,d] = attnW · V, gated, f16 ----------------
// Per head: C[i, (n,d)] = sum_j attnW[h][i][j] * Vt[h][n*32+d][j]
__global__ void __launch_bounds__(256)
k_attn_out(const _Float16* __restrict__ attnW, const _Float16* __restrict__ Vt,
           const _Float16* __restrict__ Gate, _Float16* __restrict__ Gbuf)
{
  __shared__ _Float16 GT[2][64][40];
  const int h   = blockIdx.z;
  const int i0  = blockIdx.y * 64;
  const int nd0 = blockIdx.x * 64;
  const int t = threadIdx.x, w = t >> 5, lane = t & 31;
  const int mi = w & 3, njb = (w >> 2) * 2;
  const int nn = lane & 15, sel = lane >> 4;

  const _Float16* Ah = attnW + ((size_t)h * L + i0 + mi * 16) * L;
  const _Float16* B0 = Vt + (size_t)h * KC * L + (size_t)(nd0 + njb * 16) * L;
  const _Float16* B1 = B0 + (size_t)16 * L;

  v8f acc0 = {}, acc1 = {};
#pragma unroll
  for (int ks = 0; ks < L / 32; ++ks) {
    const int c0 = ks * 32;
    v16h a  = load_frag_a_g(Ah + c0, L, lane);
    v16h b0 = load_frag_b_g(B0 + c0, L, lane);
    acc0 = wmma16(a, b0, acc0);
    v16h b1 = load_frag_b_g(B1 + c0, L, lane);
    acc1 = wmma16(a, b1, acc1);
  }

  const int n_base = nd0 >> 5;
#pragma unroll
  for (int u = 0; u < 2; ++u) {
    const int ndloc = (njb + u) * 16;
    const int nloc = ndloc >> 5;             // 0 or 1
    const int d = (ndloc & 31) + nn;         // 0..31
    const int ng = n_base + nloc;
    v8f acc = u ? acc1 : acc0;
#pragma unroll
    for (int v = 0; v < 8; ++v) {
      const int il = mi * 16 + v + 8 * sel;
      const int ig = i0 + il;
      float gv = (float)ldg_h(Gate + ((size_t)ng * L + ig) * DP + h * DH + d);
      GT[nloc][il][d] = (_Float16)(acc[v] * gv);
    }
  }
  __syncthreads();

  // coalesced store: Gbuf[(n*L + i)*128 + h*32 + d]
  {
    const int rowl = t >> 1, half = t & 1;
    const int a = rowl >> 6, il = rowl & 63;
    union { _Float16 f[16]; u32x4 q[2]; } buf;
#pragma unroll
    for (int c = 0; c < 16; ++c) buf.f[c] = GT[a][il][half * 16 + c];
    _Float16* dst = Gbuf + ((size_t)(n_base + a) * L + i0 + il) * DP + h * DH + half * 16;
    stg_u4(dst, buf.q[0]);
    stg_u4(dst + 8, buf.q[1]);
  }
}

// ---------------- kernel 5: Wo projection + bo, undo transpose ----------------
__global__ void __launch_bounds__(256)
k_out_proj(const _Float16* __restrict__ Gbuf, const _Float16* __restrict__ wo,
           const float* __restrict__ bo, float* __restrict__ out)
{
  const int r0 = blockIdx.x * 64;      // rows r = n*L + i
  const int t = threadIdx.x, w = t >> 5, lane = t & 31;
  const int mi = w & 3, nh = w >> 2;
  const int nn = lane & 15, sel = lane >> 4;
  const int n = r0 / L;
  const int ib = r0 - n * L;
  const _Float16* Abase = Gbuf + (size_t)(r0 + mi * 16) * DP;

#pragma unroll
  for (int tt = 0; tt < 4; ++tt) {
    const int nt = nh * 4 + tt;
    v8f acc = {};
#pragma unroll
    for (int ks = 0; ks < 4; ++ks) {
      v16h a  = load_frag_a_g(Abase + ks * 32, DP, lane);
      v16h bf = load_frag_b_g(wo + (size_t)nt * 16 * DP + ks * 32, DP, lane);
      acc = wmma16(a, bf, acc);
    }
    const int o = nt * 16 + nn;
    const float bov = ldg_f(bo + o);
#pragma unroll
    for (int v = 0; v < 8; ++v) {
      const int i = ib + mi * 16 + v + 8 * sel;
      stg_f(out + ((size_t)i * L + n) * DP + o, acc[v] + bov);  // final transpose (0,2,1,3)
    }
  }
}

extern "C" void kernel_launch(void* const* d_in, const int* in_sizes, int n_in,
                              void* d_out, int out_size, void* d_ws, size_t ws_size,
                              hipStream_t stream)
{
  const float* pair = (const float*)d_in[0];
  const float* bias = (const float*)d_in[1];
  const float* lnpg = (const float*)d_in[2];
  const float* lnpb = (const float*)d_in[3];
  const float* lnbg = (const float*)d_in[4];
  const float* lnbb = (const float*)d_in[5];
  const float* Wq   = (const float*)d_in[6];
  const float* Wk   = (const float*)d_in[7];
  const float* Wv   = (const float*)d_in[8];
  const float* Wb   = (const float*)d_in[9];
  const float* Wg   = (const float*)d_in[10];
  const float* bg   = (const float*)d_in[11];
  const float* Wo   = (const float*)d_in[12];
  const float* bo   = (const float*)d_in[13];
  float* out = (float*)d_out;

  char* p = (char*)d_ws;
  auto alloc = [&](size_t bytes) -> char* {
    char* r = p; p += (bytes + 255) & ~(size_t)255; return r;
  };

  const size_t NW = (size_t)DP * DP;          // 16384
  _Float16* wq16 = (_Float16*)alloc(NW * 2);
  _Float16* wk16 = (_Float16*)alloc(NW * 2);
  _Float16* wv16 = (_Float16*)alloc(NW * 2);
  _Float16* wg16 = (_Float16*)alloc(NW * 2);
  _Float16* wo16 = (_Float16*)alloc(NW * 2);
  _Float16* wb16 = (_Float16*)alloc((size_t)16 * DP * 2);
  _Float16* Qt   = (_Float16*)alloc((size_t)H * L * L * DH * 2);
  _Float16* Kt   = (_Float16*)alloc((size_t)H * L * L * DH * 2);
  _Float16* Vt   = (_Float16*)alloc((size_t)H * L * L * DH * 2);
  _Float16* Gate = (_Float16*)alloc((size_t)L * L * DP * 2);
  float*    biasP= (float*)   alloc((size_t)H * L * L * 4);
  float*    attnS= (float*)   alloc((size_t)H * L * L * 4);
  _Float16* attnW= (_Float16*)alloc((size_t)H * L * L * 2);
  _Float16* Gbuf = (_Float16*)alloc((size_t)L * L * DP * 2);

  k_f2h<<<(int)(NW / 256), 256, 0, stream>>>(Wq, wq16, (int)NW);
  k_f2h<<<(int)(NW / 256), 256, 0, stream>>>(Wk, wk16, (int)NW);
  k_f2h<<<(int)(NW / 256), 256, 0, stream>>>(Wv, wv16, (int)NW);
  k_f2h<<<(int)(NW / 256), 256, 0, stream>>>(Wg, wg16, (int)NW);
  k_f2h<<<(int)(NW / 256), 256, 0, stream>>>(Wo, wo16, (int)NW);
  k_wbpad<<<8, 256, 0, stream>>>(Wb, wb16);

  k_ln_proj_pair<<<L * 6, 256, 0, stream>>>(pair, lnpg, lnpb,
                                            wq16, wk16, wv16, wg16, bg,
                                            Qt, Kt, Vt, Gate);
  k_ln_proj_bias<<<L * 6, 256, 0, stream>>>(bias, lnbg, lnbb, wb16, biasP);
  k_scores<<<dim3(36, H), 256, 0, stream>>>(Qt, Kt, biasP, attnS);
  k_softmax<<<H * L, 128, 0, stream>>>(attnS, attnW);
  k_attn_out<<<dim3(192, 6, H), 256, 0, stream>>>(attnW, Vt, Gate, Gbuf);
  k_out_proj<<<L * 6, 256, 0, stream>>>(Gbuf, wo16, bo, out);
}
